// LISTA_22333829939512
// MI455X (gfx1250) — matrix operand order repaired and checked by reading, not compiled, not run on hardware
//
#include <hip/hip_runtime.h>
#include <hip/hip_bf16.h>
#include <stdint.h>

// ---------------------------------------------------------------------------
// LISTA fused kernel for gfx1250 (MI455X).
//   B  = We(256x64)  @ X(64x65536)      per batch
//   L0 = softshrink(B, th0)
//   Li+1 = softshrink(B + S_i(256x256) @ Li, th_{i+1})
// Compute-bound (112 GFLOP vs ~14us of HBM traffic) -> bf16 WMMA, f32 accum.
// Weights pre-packed into A-fragment layout in d_ws (L2-resident, shared);
// x / out streamed nontemporally so L2 stays dedicated to the weights.
// ---------------------------------------------------------------------------

typedef __attribute__((ext_vector_type(16))) __bf16 v16bf;
typedef __attribute__((ext_vector_type(2)))  __bf16 v2bf;
typedef __attribute__((ext_vector_type(8)))  float  v8f;
typedef __attribute__((ext_vector_type(4)))  float  v4f;

#define NPIX   64            // pixels per workgroup tile
#define HW     65536         // 256*256 pixels per (b, channel) plane
#define DICT   256
#define INCH   64
#define WE_DW  (16*2*32*8)   // packed We fragments: 8192 dwords
#define S_DW   (16*8*32*8)   // packed S_i fragments: 32768 dwords each

__device__ __forceinline__ unsigned short f2bf(float f) {
  unsigned int u = __float_as_uint(f);
  u += 0x7FFFu + ((u >> 16) & 1u);      // round-to-nearest-even
  return (unsigned short)(u >> 16);
}
__device__ __forceinline__ unsigned int pk2(float lo, float hi) {
#if __has_builtin(__builtin_amdgcn_cvt_pk_bf16_f32)
  v2bf p = __builtin_amdgcn_cvt_pk_bf16_f32(lo, hi);
  return __builtin_bit_cast(unsigned int, p);
#else
  return (unsigned int)f2bf(lo) | ((unsigned int)f2bf(hi) << 16);
#endif
}
// softshrink(x) = copysign(max(|x| - th, 0), x)  -> v_sub + v_max + v_bfi
__device__ __forceinline__ float sshrink(float x, float th) {
  float m = __builtin_fmaxf(__builtin_fabsf(x) - th, 0.0f);
  return __builtin_copysignf(m, x);
}

// Byte offset of bf16 element (k, n) inside a B-fragment staging region.
// 16-bit B 32x16 layout (contiguous k, not the ISA table order): lanes 0-15 hold K=0..15
// (VGPR j = pair K=2j,2j+1), lanes 16-31 hold K=16..31. Each (kblock, half,
// n) owns 32 contiguous bytes -> 2x ds_load_b128 per fragment.
__device__ __forceinline__ int boff(int k, int n) {
  int kb  = k >> 5;
  int kin = k & 31;
  int hi  = kin >> 4;
  int j   = (kin & 15) >> 1;
  return ((kb * 2 + hi) * NPIX + n) * 32 + j * 4 + (kin & 1) * 2;
}

// ---------------------------------------------------------------------------
// Pre-pack We and S into bf16 A-fragment layout (ISA 7.12.2, 16-bit A 16x32):
// lane L (0-15): row M=L, VGPR j -> K=2j,2j+1 (contiguous K=0..15, not the ISA table order)
// lane L (16-31): same rows, VGPR j -> K=16+2j,17+2j (K=16..31).
// ---------------------------------------------------------------------------
__global__ void lista_pack(const float* __restrict__ We,
                           const float* __restrict__ S,
                           unsigned int* __restrict__ wp) {
  int idx = blockIdx.x * blockDim.x + threadIdx.x;
  if (idx < WE_DW) {
    int j  = idx & 7;
    int L  = (idx >> 3) & 31;
    int kb = (idx >> 8) & 1;
    int mt = idx >> 9;
    int m    = mt * 16 + (L & 15);
    int koff = ((L >> 4) * 16) + 2 * j;
    int k    = kb * 32 + koff;
    wp[idx] = pk2(We[m * INCH + k], We[m * INCH + k + 1]);
  } else if (idx < WE_DW + 3 * S_DW) {
    int r = idx - WE_DW;
    int i = r / S_DW;
    int q = r % S_DW;
    int j  = q & 7;
    int L  = (q >> 3) & 31;
    int kb = (q >> 8) & 7;
    int mt = q >> 11;
    int m    = mt * 16 + (L & 15);
    int koff = ((L >> 4) * 16) + 2 * j;
    int k    = kb * 32 + koff;
    const float* W = S + (size_t)i * DICT * DICT;
    wp[idx] = pk2(W[m * DICT + k], W[m * DICT + k + 1]);
  }
}

// Store softshrunk accumulator tile into the logits LDS region as packed bf16.
// C/D layout: lane 0-15 VGPR v -> M=v ; lane 16-31 -> M=v+8; N = lane&15.
// Pairs (m, m+1) live in adjacent VGPRs of the same lane -> one b128 store.
__device__ __forceinline__ void store_logits(unsigned char* llds, v8f v,
                                             int mt, int nt, int lane, float th) {
  int hiL = lane >> 4, nn = lane & 15;
  int mb  = mt * 16 + hiL * 8;
  int kbk = mb >> 5, kin = mb & 31;
  int off = ((kbk * 2 + (kin >> 4)) * NPIX + nt * 16 + nn) * 32 + ((kin & 8) ? 16 : 0);
  uint4 p;
  p.x = pk2(sshrink(v[0], th), sshrink(v[1], th));
  p.y = pk2(sshrink(v[2], th), sshrink(v[3], th));
  p.z = pk2(sshrink(v[4], th), sshrink(v[5], th));
  p.w = pk2(sshrink(v[6], th), sshrink(v[7], th));
  *(uint4*)(llds + off) = p;
}

__device__ __forceinline__ void store_out(float* __restrict__ out, v8f v,
                                          int b, int hw0, int mt, int nt,
                                          int lane, float th) {
  int hiL = lane >> 4, nn = lane & 15;
  int mb  = mt * 16 + hiL * 8;
  float* op = out + ((size_t)b * DICT + mb) * HW + hw0 + nt * 16 + nn;
#pragma unroll
  for (int r = 0; r < 8; ++r)
    __builtin_nontemporal_store(sshrink(v[r], th), op + (size_t)r * HW);
}

__global__ __launch_bounds__(256, 1)
void lista_main(const float* __restrict__ x,
                const unsigned int* __restrict__ wp,
                const float* __restrict__ thetas,
                float* __restrict__ out) {
  __shared__ __align__(16) unsigned char smem[8192 + 32768];
  unsigned char* xlds = smem;          // 64ch x 64pix bf16, B-frag layout
  unsigned char* llds = smem + 8192;   // 256 x 64 logits bf16, B-frag layout

  const int t    = threadIdx.x;
  const int lane = t & 31;
  const int wid  = t >> 5;
  const int hiL  = lane >> 4;
  const int nn   = lane & 15;
  const int mt0  = wid * 2;            // 8 waves x 2 mtiles = 256 dict rows

  const int tile = blockIdx.x;         // 4096 tiles
  const int b    = tile >> 10;         // 1024 tiles per batch image
  const int hw0  = (tile & 1023) * NPIX;

  const float th0 = thetas[0];
  const float ths[3] = { thetas[1], thetas[2], thetas[3] };

  // ---- stage x tile: each thread loads 16 contiguous f32 of one channel ----
  {
    const int c = t >> 2;
    const int g = (t & 3) * 16;
    const v4f* xr4 = (const v4f*)(x + ((size_t)b * INCH + c) * HW + hw0 + g);
#pragma unroll
    for (int q = 0; q < 4; ++q) {
      v4f v = __builtin_nontemporal_load(xr4 + q);   // read-once stream
      *(unsigned short*)(xlds + boff(c, g + 4 * q + 0)) = f2bf(v.x);
      *(unsigned short*)(xlds + boff(c, g + 4 * q + 1)) = f2bf(v.y);
      *(unsigned short*)(xlds + boff(c, g + 4 * q + 2)) = f2bf(v.z);
      *(unsigned short*)(xlds + boff(c, g + 4 * q + 3)) = f2bf(v.w);
    }
  }
  __syncthreads();

  // ---- GEMM1: B = We @ X  (K = 64 -> 2 k-blocks) ----
  v8f acc0[4], acc1[4];
#pragma unroll
  for (int nt = 0; nt < 4; ++nt) {
    acc0[nt] = (v8f){0.f, 0.f, 0.f, 0.f, 0.f, 0.f, 0.f, 0.f};
    acc1[nt] = (v8f){0.f, 0.f, 0.f, 0.f, 0.f, 0.f, 0.f, 0.f};
  }
#pragma unroll
  for (int kb = 0; kb < 2; ++kb) {
    v16bf A0 = *(const v16bf*)(wp + (((mt0    ) * 2 + kb) * 32 + lane) * 8);
    v16bf A1 = *(const v16bf*)(wp + (((mt0 + 1) * 2 + kb) * 32 + lane) * 8);
#pragma unroll
    for (int nt = 0; nt < 4; ++nt) {
      v16bf B = *(const v16bf*)(xlds + ((kb * 2 + hiL) * NPIX + nt * 16 + nn) * 32);
      acc0[nt] = __builtin_amdgcn_wmma_f32_16x16x32_bf16(false, A0, false, B,
                                                         (short)0, acc0[nt], false, false);
      acc1[nt] = __builtin_amdgcn_wmma_f32_16x16x32_bf16(false, A1, false, B,
                                                         (short)0, acc1[nt], false, false);
    }
  }

  // raw B stays resident in registers for the recurrence
  v8f b0[4], b1[4];
#pragma unroll
  for (int nt = 0; nt < 4; ++nt) { b0[nt] = acc0[nt]; b1[nt] = acc1[nt]; }

  // logits_0 = softshrink(B, th0) -> LDS (packed bf16, B-frag layout)
#pragma unroll
  for (int nt = 0; nt < 4; ++nt) {
    store_logits(llds, acc0[nt], mt0,     nt, lane, th0);
    store_logits(llds, acc1[nt], mt0 + 1, nt, lane, th0);
  }
  __syncthreads();

  // ---- 3 LISTA iterations: acc = B + S_i @ logits ----
#pragma unroll
  for (int it = 0; it < 3; ++it) {
    const unsigned int* sA = wp + WE_DW + it * S_DW;
    __builtin_prefetch(sA, 0, 1);   // global_prefetch_b8: pull S_i toward WGP
#pragma unroll
    for (int kb = 0; kb < 8; ++kb) {
      v16bf A0 = *(const v16bf*)(sA + (((mt0    ) * 8 + kb) * 32 + lane) * 8);
      v16bf A1 = *(const v16bf*)(sA + (((mt0 + 1) * 8 + kb) * 32 + lane) * 8);
#pragma unroll
      for (int nt = 0; nt < 4; ++nt) {
        v16bf B = *(const v16bf*)(llds + ((kb * 2 + hiL) * NPIX + nt * 16 + nn) * 32);
        // kb==0 folds at compile time: WMMA C-input = raw Breg (no v_mov copies)
        v8f c0 = (kb == 0) ? b0[nt] : acc0[nt];
        v8f c1 = (kb == 0) ? b1[nt] : acc1[nt];
        acc0[nt] = __builtin_amdgcn_wmma_f32_16x16x32_bf16(false, A0, false, B,
                                                           (short)0, c0, false, false);
        acc1[nt] = __builtin_amdgcn_wmma_f32_16x16x32_bf16(false, A1, false, B,
                                                           (short)0, c1, false, false);
      }
    }
    const float th = ths[it];
    if (it < 2) {
      __syncthreads();   // all B-frag reads done before overwriting logits
#pragma unroll
      for (int nt = 0; nt < 4; ++nt) {
        store_logits(llds, acc0[nt], mt0,     nt, lane, th);
        store_logits(llds, acc1[nt], mt0 + 1, nt, lane, th);
      }
      __syncthreads();
    } else {
#pragma unroll
      for (int nt = 0; nt < 4; ++nt) {
        store_out(out, acc0[nt], b, hw0, mt0,     nt, lane, th);
        store_out(out, acc1[nt], b, hw0, mt0 + 1, nt, lane, th);
      }
    }
  }
}

extern "C" void kernel_launch(void* const* d_in, const int* in_sizes, int n_in,
                              void* d_out, int out_size, void* d_ws, size_t ws_size,
                              hipStream_t stream) {
  const float* x  = (const float*)d_in[0];   // (4, 64, 256, 256) f32
  const float* We = (const float*)d_in[1];   // (256, 64) f32
  const float* S  = (const float*)d_in[2];   // (3, 256, 256) f32
  const float* th = (const float*)d_in[3];   // (4,) f32
  unsigned int* wp = (unsigned int*)d_ws;    // needs 425,984 bytes

  const int packN = WE_DW + 3 * S_DW;        // 106,496 dwords
  lista_pack<<<(packN + 255) / 256, 256, 0, stream>>>(We, S, wp);
  lista_main<<<4096, 256, 0, stream>>>(x, wp, th, (float*)d_out);
}
